// HeteroGNN_16338055594512
// MI455X (gfx1250) — matrix operand order repaired and checked by reading, not compiled, not run on hardware
//
#include <hip/hip_runtime.h>

typedef __attribute__((ext_vector_type(16))) _Float16 v16h;
typedef __attribute__((ext_vector_type(8)))  _Float16 v8h;
typedef __attribute__((ext_vector_type(8)))  float    v8f;

#define NB 100000
#define NG 30000
#define NL 60000
#define HD 64

#define GF_PRERELU   1
#define GF_ACCUM     2
#define GF_MID_RELU  4
#define GF_POST_RELU 8
#define GF_HAS_MM2   16
#define GF_HAS_BN    32
#define GF_ADD_AGG   64

#define GINE_COMMON (GF_ADD_AGG | GF_HAS_MM2 | GF_HAS_BN | GF_MID_RELU | GF_POST_RELU)
#define GINE_ACC    (GINE_COMMON | GF_ACCUM)
#define GINE_WR     (GINE_COMMON)
#define HEAD_FLAGS  (GF_PRERELU)

// ---------------- elementwise helpers ----------------
__global__ void zero_f(float* p, int n) {
    int i = blockIdx.x * 256 + threadIdx.x;
    if (i < n) p[i] = 0.f;
}

__global__ void post_conv(const float* __restrict__ h, float s,
                          float* __restrict__ x, float* __restrict__ jk, int n) {
    int i = blockIdx.x * 256 + threadIdx.x;
    if (i < n) {
        float v = h[i] * s;
        v = v > 0.f ? v : 0.2f * v;   // leaky_relu(0.2)
        x[i] = v;
        jk[i] += v;
    }
}

__global__ void scale_copy(const float* __restrict__ a, float s,
                           float* __restrict__ o, int n) {
    int i = blockIdx.x * 256 + threadIdx.x;
    if (i < n) o[i] = a[i] * s;
}

// input projection: X[n,K] @ W[K,64] + b  (K = 4 or 8; memory bound, VALU fine)
__global__ void lin_in(const float* __restrict__ X, const float* __restrict__ W,
                       const float* __restrict__ b, float* __restrict__ o,
                       int n, int K) {
    int i = blockIdx.x * 256 + threadIdx.x;
    if (i >= n * HD) return;
    int row = i >> 6, j = i & 63;
    float acc = b[j];
    const float* xr = X + (size_t)row * K;
    for (int k = 0; k < K; ++k) acc += xr[k] * W[k * HD + j];
    o[i] = acc;
}

// output head: X[n,64] @ W[64,C] + b   (C = 2 or 4)
__global__ void lin_out(const float* __restrict__ X, const float* __restrict__ W,
                        const float* __restrict__ b, float* __restrict__ o,
                        int n, int C) {
    int i = blockIdx.x * 256 + threadIdx.x;
    if (i >= n) return;
    float acc[4];
    for (int c = 0; c < 4; ++c) acc[c] = (c < C) ? b[c] : 0.f;
    const float* xr = X + (size_t)i * HD;
    for (int k = 0; k < HD; ++k) {
        float x = xr[k];
        for (int c = 0; c < 4; ++c)
            if (c < C) acc[c] += x * W[k * C + c];
    }
    for (int c = 0; c < C; ++c) o[(size_t)i * C + c] = acc[c];
}

// ---------------- fused edge message + scatter-add ----------------
// msg = relu(x_src[src] + attr@We + be); agg[dst] += msg   (fp32 HW atomics in L2)
// 16 threads per edge, 4 features (float4) per thread.
__global__ __launch_bounds__(256) void edge_msg(
    const float* __restrict__ Xs, const int* __restrict__ src,
    const int* __restrict__ dst, const float* __restrict__ attr,
    const float* __restrict__ We, const float* __restrict__ be,
    float* __restrict__ agg, int nE)
{
    int t = threadIdx.x;
    int e = blockIdx.x * 16 + (t >> 4);
    if (e >= nE) return;
    int fg = (t & 15) * 4;
    const float4 av = *(const float4*)(attr + (size_t)e * 4);
    int s = src[e], d = dst[e];
    const float4 xv = *(const float4*)(Xs + (size_t)s * HD + fg);
    float* ar = agg + (size_t)d * HD + fg;
    float xr[4] = {xv.x, xv.y, xv.z, xv.w};
#pragma unroll
    for (int c = 0; c < 4; ++c) {
        int j = fg + c;
        float ev = be[j] + av.x * We[j] + av.y * We[HD + j]
                         + av.z * We[2 * HD + j] + av.w * We[3 * HD + j];
        float m = xr[c] + ev;
        m = m > 0.f ? m : 0.f;
        unsafeAtomicAdd(ar + c, m);
    }
}

// ---------------- fused WMMA node update (flags are compile-time) ----------------
// Per 16-row tile (one wave):  in = [pre-relu?](X (+AGG))
//   h1 = in @ W1 + b1 ; [BN] ; [relu] ; [h2 = h1 @ W2 + b2 ; [relu]]
//   Out = (ACCUM ? Out + r : r)
// v_wmma_f32_16x16x32_f16, f32 accumulate; weights staged transposed in LDS.
template <int FLAGS>
__global__ __launch_bounds__(256) void gine_wmma(
    const float* __restrict__ X, const float* __restrict__ AGG,
    const float* __restrict__ W1, const float* __restrict__ B1,
    const float* __restrict__ BNG, const float* __restrict__ BNB,
    const float* __restrict__ BNM, const float* __restrict__ BNV,
    const float* __restrict__ W2, const float* __restrict__ B2,
    float* __restrict__ Out, int n)
{
    constexpr bool has2   = (FLAGS & GF_HAS_MM2) != 0;
    constexpr bool hasbn  = (FLAGS & GF_HAS_BN) != 0;
    constexpr bool addagg = (FLAGS & GF_ADD_AGG) != 0;
    constexpr bool prer   = (FLAGS & GF_PRERELU) != 0;
    constexpr bool midr   = (FLAGS & GF_MID_RELU) != 0;
    constexpr bool postr  = (FLAGS & GF_POST_RELU) != 0;
    constexpr bool accum  = (FLAGS & GF_ACCUM) != 0;

    __shared__ __align__(32) _Float16 Wt1[HD * HD];
    __shared__ __align__(32) _Float16 Wt2[has2 ? HD * HD : 16];
    __shared__ __align__(32) _Float16 Abuf[8][16 * HD];
    __shared__ float b1s[HD], b2s[HD], bsc[HD], bsh[HD];

    int t = threadIdx.x;
    for (int i = t; i < HD * HD; i += 256) {
        int k = i >> 6, nn = i & 63;
        Wt1[nn * HD + k] = (_Float16)W1[i];
        if constexpr (has2) Wt2[nn * HD + k] = (_Float16)W2[i];
    }
    if (t < HD) {
        b1s[t] = B1[t];
        if constexpr (has2) b2s[t] = B2[t];
        if constexpr (hasbn) {
            float s = BNG[t] * rsqrtf(BNV[t] + 1e-5f);
            bsc[t] = s;
            bsh[t] = BNB[t] - BNM[t] * s;
        }
    }
    __syncthreads();

    int wave = t >> 5, lane = t & 31;
    int m16 = lane & 15, hh = lane >> 4;
    int tile = blockIdx.x * 8 + wave;
    int row0 = tile * 16;
    if (row0 >= n) return;
    _Float16* Ab = Abuf[wave];

    // stage input tile (row-major f16) : coalesced 128B global reads
    for (int i = 0; i < 32; ++i) {
        int idx = i * 32 + lane;        // 0..1023
        int r = idx >> 6, c = idx & 63;
        int row = row0 + r;
        float v = 0.f;
        if (row < n) {
            v = X[(size_t)row * HD + c];
            if constexpr (addagg) v += AGG[(size_t)row * HD + c];
            if constexpr (prer) v = v > 0.f ? v : 0.f;
        }
        Ab[idx] = (_Float16)v;
    }

    typedef union { v16h v; v8h h[2]; } vu;
    v8f acc[4];

    // A-fragment (ISA 16-bit A 16x32 layout): lane(m16,hh):
    //   elems 0..7  -> K = kk*32 + 8*hh + (0..7)
    //   elems 8..15 -> K = kk*32 + 16 + 8*hh + (0..7)
    // B-fragment from Wt[n][k]: lane(n=m16,hh): same K layout as A
    auto do_mm = [&](const _Float16* Wt) {
        vu a0, a1;
        const _Float16* ap = Ab + m16 * HD;
        a0.h[0] = *(const v8h*)(ap + 8 * hh);
        a0.h[1] = *(const v8h*)(ap + 16 + 8 * hh);
        a1.h[0] = *(const v8h*)(ap + 32 + 8 * hh);
        a1.h[1] = *(const v8h*)(ap + 32 + 16 + 8 * hh);
#pragma unroll
        for (int j = 0; j < 4; ++j) {
            const _Float16* wp = Wt + (j * 16 + m16) * HD;
            vu b0, b1v;
            b0.h[0]  = *(const v8h*)(wp + 8 * hh);
            b0.h[1]  = *(const v8h*)(wp + 16 + 8 * hh);
            b1v.h[0] = *(const v8h*)(wp + 32 + 8 * hh);
            b1v.h[1] = *(const v8h*)(wp + 32 + 16 + 8 * hh);
            v8f c = {};
            c = __builtin_amdgcn_wmma_f32_16x16x32_f16(false, a0.v, false, b0.v,
                                                       (short)0, c, false, false);
            c = __builtin_amdgcn_wmma_f32_16x16x32_f16(false, a1.v, false, b1v.v,
                                                       (short)0, c, false, false);
            acc[j] = c;
        }
    };

    do_mm(Wt1);
    // epilogue mm1 ; D layout: VGPR r -> row = r + 8*hh, col = j*16 + m16
#pragma unroll
    for (int j = 0; j < 4; ++j) {
        int col = j * 16 + m16;
        float bb = b1s[col];
        float s = hasbn ? bsc[col] : 1.f;
        float o = hasbn ? bsh[col] : 0.f;
#pragma unroll
        for (int r = 0; r < 8; ++r) {
            float v = acc[j][r] + bb;
            if constexpr (hasbn) v = v * s + o;
            if constexpr (midr) v = v > 0.f ? v : 0.f;
            if constexpr (has2) {
                Ab[(r + 8 * hh) * HD + col] = (_Float16)v;  // restage as A for mm2
            } else {
                int row = row0 + r + 8 * hh;
                if (row < n) {
                    float* op = Out + (size_t)row * HD + col;
                    *op = accum ? (*op + v) : v;
                }
            }
        }
    }
    if constexpr (has2) {
        do_mm(Wt2);
#pragma unroll
        for (int j = 0; j < 4; ++j) {
            int col = j * 16 + m16;
            float bb = b2s[col];
#pragma unroll
            for (int r = 0; r < 8; ++r) {
                float v = acc[j][r] + bb;
                if constexpr (postr) v = v > 0.f ? v : 0.f;
                int row = row0 + r + 8 * hh;
                if (row < n) {
                    float* op = Out + (size_t)row * HD + col;
                    *op = accum ? (*op + v) : v;
                }
            }
        }
    }
}

// ---------------- host orchestration ----------------
extern "C" void kernel_launch(void* const* d_in, const int* in_sizes, int n_in,
                              void* d_out, int out_size, void* d_ws, size_t ws_size,
                              hipStream_t stream)
{
    (void)in_sizes; (void)n_in; (void)out_size; (void)ws_size;
    auto F = [&](int i) { return (const float*)d_in[i]; };
    auto I = [&](int i) { return (const int*)d_in[i]; };

    const float* x_bus  = F(0);
    const float* x_gen  = F(1);
    const float* x_load = F(2);

    // params flatten (sorted keys): convs[3][bb,bg,bl,gb,lb][bn_b,bn_g,bn_m,bn_v,eW,eb,W1,b1,W2,b2]
    const int PB = 18;        // convs base
    const int IB = 168;       // inp base
    const int LB = 174;       // lins base

    const size_t SB = (size_t)NB * HD, SG = (size_t)NG * HD, SL = (size_t)NL * HD;
    float* ws = (float*)d_ws;
    float* xb = ws;            float* xg = xb + SB;  float* xl = xg + SG;
    float* hb = xl + SL;       float* hg = hb + SB;  float* hl = hg + SG;
    float* jb = hl + SL;       float* jg = jb + SB;  float* jl = jg + SG;
    float* agg = jl + SL;      // NB*64, reused for every edge type

    dim3 B(256);
    auto zero = [&](float* p, size_t nf) {
        zero_f<<<dim3((unsigned)((nf + 255) / 256)), B, 0, stream>>>(p, (int)nf);
    };
    auto emsg = [&](const float* Xs, int si, int di, int ai, int l, int tE,
                    float* AG, int nE) {
        int base = PB + l * 50 + tE * 10;
        edge_msg<<<dim3((nE + 15) / 16), B, 0, stream>>>(
            Xs, I(si), I(di), F(ai), F(base + 4), F(base + 5), AG, nE);
    };
    auto gine = [&](const float* X, const float* AG, int l, int tE,
                    float* Outp, int nN, bool accum) {
        int base = PB + l * 50 + tE * 10;
        int tiles = (nN + 15) / 16;
        dim3 G((tiles + 7) / 8);
        if (accum)
            gine_wmma<GINE_ACC><<<G, B, 0, stream>>>(
                X, AG, F(base + 6), F(base + 7),
                F(base + 1), F(base + 0), F(base + 2), F(base + 3),
                F(base + 8), F(base + 9), Outp, nN);
        else
            gine_wmma<GINE_WR><<<G, B, 0, stream>>>(
                X, AG, F(base + 6), F(base + 7),
                F(base + 1), F(base + 0), F(base + 2), F(base + 3),
                F(base + 8), F(base + 9), Outp, nN);
    };
    auto headlin = [&](const float* X, int wi, int bi, float* Outp, int nN) {
        int tiles = (nN + 15) / 16;
        gine_wmma<HEAD_FLAGS><<<dim3((tiles + 7) / 8), B, 0, stream>>>(
            X, nullptr, F(wi), F(bi),
            nullptr, nullptr, nullptr, nullptr, nullptr, nullptr,
            Outp, nN);
    };

    // input projections
    lin_in<<<dim3((unsigned)((SB + 255) / 256)), B, 0, stream>>>(x_bus,  F(IB+0), F(IB+1), xb, NB, 8);
    lin_in<<<dim3((unsigned)((SG + 255) / 256)), B, 0, stream>>>(x_gen,  F(IB+2), F(IB+3), xg, NG, 4);
    lin_in<<<dim3((unsigned)((SL + 255) / 256)), B, 0, stream>>>(x_load, F(IB+4), F(IB+5), xl, NL, 4);
    zero(jb, SB); zero(jg, SG); zero(jl, SL);

    for (int l = 0; l < 3; ++l) {
        // --- bus destination: bb (t=0), gb (t=3), lb (t=4), mean later ---
        zero(hb, SB);
        zero(agg, SB); emsg(xb, 3, 4, 5, l, 0, agg, 1000000);   gine(xb, agg, l, 0, hb, NB, true);
        zero(agg, SB); emsg(xg, 6, 7, 8, l, 3, agg, 300000);    gine(xb, agg, l, 3, hb, NB, true);
        zero(agg, SB); emsg(xl, 12, 13, 14, l, 4, agg, 600000); gine(xb, agg, l, 4, hb, NB, true);
        // --- gen destination: bg (t=1) ---
        zero(agg, SG); emsg(xb, 9, 10, 11, l, 1, agg, 300000);  gine(xg, agg, l, 1, hg, NG, false);
        // --- load destination: bl (t=2) ---
        zero(agg, SL); emsg(xb, 15, 16, 17, l, 2, agg, 600000); gine(xl, agg, l, 2, hl, NL, false);
        // x = lrelu(h [/3 for bus]) ; jk += x
        post_conv<<<dim3((unsigned)((SB + 255) / 256)), B, 0, stream>>>(hb, 1.f / 3.f, xb, jb, (int)SB);
        post_conv<<<dim3((unsigned)((SG + 255) / 256)), B, 0, stream>>>(hg, 1.f, xg, jg, (int)SG);
        post_conv<<<dim3((unsigned)((SL + 255) / 256)), B, 0, stream>>>(hl, 1.f, xl, jl, (int)SL);
    }

    // jumping knowledge mean
    scale_copy<<<dim3((unsigned)((SB + 255) / 256)), B, 0, stream>>>(jb, 1.f / 3.f, xb, (int)SB);
    scale_copy<<<dim3((unsigned)((SG + 255) / 256)), B, 0, stream>>>(jg, 1.f / 3.f, xg, (int)SG);
    scale_copy<<<dim3((unsigned)((SL + 255) / 256)), B, 0, stream>>>(jl, 1.f / 3.f, xl, (int)SL);

    // MLP head: two 64x64 linears with pre-relu, then final 64->C
    headlin(xb, LB + 0, LB + 1, hb, NB);  headlin(hb, LB + 6, LB + 7,  xb, NB);
    headlin(xg, LB + 2, LB + 3, hg, NG);  headlin(hg, LB + 8, LB + 9,  xg, NG);
    headlin(xl, LB + 4, LB + 5, hl, NL);  headlin(hl, LB + 10, LB + 11, xl, NL);

    float* ob = (float*)d_out;
    lin_out<<<dim3((NB + 255) / 256), B, 0, stream>>>(xb, F(LB + 12), F(LB + 13), ob, NB, 4);
    lin_out<<<dim3((NG + 255) / 256), B, 0, stream>>>(xg, F(LB + 14), F(LB + 15), ob + (size_t)NB * 4, NG, 2);
    lin_out<<<dim3((NL + 255) / 256), B, 0, stream>>>(xl, F(LB + 16), F(LB + 17), ob + (size_t)NB * 4 + (size_t)NG * 2, NL, 2);
}
